// DGCNN_58823872086173
// MI455X (gfx1250) — matrix operand order repaired and checked by reading, not compiled, not run on hardware
//
#include <hip/hip_runtime.h>
#include <stdint.h>

#define B_GRAPHS 256
#define NPER     256
#define NN       (B_GRAPHS*NPER)   // 65536
#define DEG      16
#define EDGES    (NN*DEG)          // 1048576
#define HID      128
#define LAYERS   3
#define DFEAT    (HID*LAYERS)      // 384
#define KSORT    30

typedef _Float16 h16;
typedef __attribute__((ext_vector_type(16))) _Float16 v16h;
typedef __attribute__((ext_vector_type(8)))  _Float16 v8h;
typedef __attribute__((ext_vector_type(8)))  float    v8f;

#define ADJ_STRIDE 264   // 256 + 8 halves pad (528B row: 4-bank rotation, 16B aligned)
#define H_STRIDE   264   // transposed h: [128 features][256 nodes + pad]

// branch-free tanh: 1 - 2/(e^{2x}+1); saturates via exp2->inf/0, no EXEC divergence
__device__ __forceinline__ float fast_tanh(float x) {
  float t = __builtin_amdgcn_exp2f(x * 2.8853900817779268f);  // e^(2x)
  return fmaf(-2.0f, __builtin_amdgcn_rcpf(t + 1.0f), 1.0f);
}

// ---------------- degree / norm prep ----------------
__global__ void k_zero(unsigned* __restrict__ p, int n) {
  int i = blockIdx.x * 256 + threadIdx.x;
  if (i < n) p[i] = 0u;
}

__global__ void k_deg(const int* __restrict__ src, const int* __restrict__ dst,
                      unsigned* __restrict__ dout, unsigned* __restrict__ din) {
  int e = blockIdx.x * 256 + threadIdx.x;
  if (e < EDGES) {
    atomicAdd(&dout[src[e]], 1u);
    atomicAdd(&din[dst[e]], 1u);
  }
}

__global__ void k_norm(const unsigned* __restrict__ dout, const unsigned* __restrict__ din,
                       float* __restrict__ ns, float* __restrict__ nd) {
  int i = blockIdx.x * 256 + threadIdx.x;
  if (i < NN) {
    ns[i] = rsqrtf(fmaxf((float)dout[i], 1.0f));
    nd[i] = rsqrtf(fmaxf((float)din[i], 1.0f));
  }
}

// ---------------- per-graph 3-layer GraphConv as dense WMMA GEMM ----------------
// One workgroup (256 threads = 8 waves) per graph.
// LDS: adj f16 [256][264] (132KB) + hT ping/pong f16 [128][264] x2 (132KB) = 264KB
__global__ __launch_bounds__(256) void k_gconv(
    const int* __restrict__ z, const int* __restrict__ src, const int* __restrict__ dst,
    const float* __restrict__ z_table, const float* __restrict__ biases,
    const float* __restrict__ norm_src, const float* __restrict__ norm_dst,
    float* __restrict__ x)
{
  extern __shared__ __align__(16) unsigned char smem[];
  h16* adj = (h16*)smem;                                     // [256][ADJ_STRIDE]
  h16* hT0 = (h16*)(smem + (size_t)256 * ADJ_STRIDE * 2);    // [128][H_STRIDE]
  h16* hT1 = hT0 + 128 * H_STRIDE;

  const int g   = blockIdx.x;
  const int tid = threadIdx.x;

  // zero adjacency (incl. pad) as dwords
  unsigned* adj32 = (unsigned*)adj;
  for (int i = tid; i < 256 * ADJ_STRIDE / 2; i += 256) adj32[i] = 0u;

  // h0 = z_table[z], stored transposed hT0[f][node] in f16
  {
    const int node = tid;
    const int zi = z[g * NPER + node];
    const float* row = z_table + (size_t)zi * HID;
    for (int f = 0; f < HID; ++f) hT0[f * H_STRIDE + node] = (h16)row[f];
  }
  __syncthreads();

  // Build normalized adjacency: adj[d][s] += norm_dst[d]*norm_src[s] per edge.
  // Packed-f16 LDS atomics handle multi-edges and the 2-halves-per-dword hazard.
  for (int e = g * NPER * DEG + tid; e < (g + 1) * NPER * DEG; e += 256) {
    const int sg = src[e], dg = dst[e];
    const int sl = sg - g * NPER, dl = dg - g * NPER;
    const float val = norm_src[sg] * norm_dst[dg];
    union { h16 h; unsigned short u; } cv; cv.h = (h16)val;
    const unsigned data = (sl & 1) ? ((unsigned)cv.u << 16) : (unsigned)cv.u;
    const unsigned addr = (unsigned)(uintptr_t)(&adj[dl * ADJ_STRIDE + (sl & ~1)]);
    asm volatile("ds_pk_add_f16 %0, %1" :: "v"(addr), "v"(data) : "memory");
  }
  asm volatile("s_wait_dscnt 0" ::: "memory");
  __syncthreads();

  // GEMM: agg[d][f] = sum_s adj[d][s] * h[s][f];  h_next = tanh(agg + bias)
  const int lane = tid & 31;
  const int w    = tid >> 5;            // wave 0..7 -> owns n-tile = w (features w*16..w*16+15)
  const int r16  = lane & 15;           // row/col within tile
  const int hi   = lane >> 4;
  const int koffA = hi * 8;             // A-matrix per-lane K pattern: {koffA..+8} u {koffA+16..+8}
  const int koffB = hi * 8;             // B-matrix per-lane K pattern: {koffB..+8} u {koffB+16..+8}
  const int f    = w * 16 + r16;        // this lane's fixed feature column

  h16* hcur = hT0;
  h16* hnxt = hT1;

  for (int l = 0; l < LAYERS; ++l) {
    const float bias = biases[l * HID + f];
    for (int mt = 0; mt < 16; ++mt) {
      v8f acc = {0.f, 0.f, 0.f, 0.f, 0.f, 0.f, 0.f, 0.f};
      const int arow = mt * 16 + r16;
      #pragma unroll
      for (int kt = 0; kt < 8; ++kt) {
        const int kbA = kt * 32 + koffA;
        const int kbB = kt * 32 + koffB;
        v8h alo = *(const v8h*)(adj + arow * ADJ_STRIDE + kbA);
        v8h ahi = *(const v8h*)(adj + arow * ADJ_STRIDE + kbA + 16);
        v8h blo = *(const v8h*)(hcur + f * H_STRIDE + kbB);
        v8h bhi = *(const v8h*)(hcur + f * H_STRIDE + kbB + 16);
        v16h a = __builtin_shufflevector(alo, ahi, 0,1,2,3,4,5,6,7,8,9,10,11,12,13,14,15);
        v16h b = __builtin_shufflevector(blo, bhi, 0,1,2,3,4,5,6,7,8,9,10,11,12,13,14,15);
        acc = __builtin_amdgcn_wmma_f32_16x16x32_f16(false, a, false, b,
                                                     (short)0, acc, false, false);
      }
      // epilogue: lane holds rows m = r + 8*hi (r=0..7) of column f
      const int col0 = mt * 16 + hi * 8;
      v8h outv;
      #pragma unroll
      for (int r = 0; r < 8; ++r) {
        const float v = fast_tanh(acc[r] + bias);
        outv[r] = (h16)v;
        x[(size_t)(g * NPER + col0 + r) * DFEAT + l * HID + f] = v;
      }
      *(v8h*)(hnxt + f * H_STRIDE + col0) = outv;   // contiguous 16B packed store
    }
    __syncthreads();
    h16* t = hcur; hcur = hnxt; hnxt = t;
  }
}

// ---------------- per-graph SortPooling + CNN/MLP head ----------------
__global__ __launch_bounds__(256) void k_head(
    const float* __restrict__ x,
    const float* __restrict__ w1, const float* __restrict__ b1,
    const float* __restrict__ w2, const float* __restrict__ b2,
    const float* __restrict__ lw1, const float* __restrict__ lb1,
    const float* __restrict__ lw2, const float* __restrict__ lb2,
    float* __restrict__ out)
{
  __shared__ float val[256];
  __shared__ int   idx[256];
  __shared__ float sbuf[512];
  __shared__ float pooled[KSORT * DFEAT];   // 46080B
  __shared__ float c1[16 * KSORT];
  __shared__ float pp[16 * 15];
  __shared__ float c2[32 * 11];
  __shared__ float hh[128];

  const int g = blockIdx.x, tid = threadIdx.x;
  const float* xg = x + (size_t)g * NPER * DFEAT;

  // per-node max over 384 features (== last element after ascending sort)
  {
    const float* row = xg + (size_t)tid * DFEAT;
    float m = row[0];
    for (int j = 1; j < DFEAT; ++j) m = fmaxf(m, row[j]);
    val[tid] = m; idx[tid] = tid;
  }
  __syncthreads();

  // bitonic sort 256: descending by value, stable tie-break by lower index (matches top_k)
  for (int k = 2; k <= 256; k <<= 1) {
    for (int j = k >> 1; j > 0; j >>= 1) {
      const int i = tid, ixj = i ^ j;
      if (ixj > i) {
        float v0 = val[i], v1v = val[ixj];
        int   a0 = idx[i], a1 = idx[ixj];
        const bool before = (v0 > v1v) || (v0 == v1v && a0 < a1);
        const bool desc = ((i & k) == 0);
        if (desc ? !before : before) { val[i] = v1v; val[ixj] = v0; idx[i] = a1; idx[ixj] = a0; }
      }
      __syncthreads();
    }
  }

  // sort each selected node's 384 features ascending (pad to 512 with +inf)
  for (int kk = 0; kk < KSORT; ++kk) {
    const int node = idx[kk];
    const float* row = xg + (size_t)node * DFEAT;
    for (int j = tid; j < 512; j += 256) sbuf[j] = (j < DFEAT) ? row[j] : 3.0e38f;
    __syncthreads();
    for (int k = 2; k <= 512; k <<= 1) {
      for (int j = k >> 1; j > 0; j >>= 1) {
        for (int i = tid; i < 512; i += 256) {
          const int ixj = i ^ j;
          if (ixj > i) {
            const float a = sbuf[i], b = sbuf[ixj];
            const bool asc = ((i & k) == 0);
            if (asc ? (a > b) : (a < b)) { sbuf[i] = b; sbuf[ixj] = a; }
          }
        }
        __syncthreads();
      }
    }
    for (int j = tid; j < DFEAT; j += 256) pooled[kk * DFEAT + j] = sbuf[j];
    __syncthreads();
  }

  // conv1: c1[o][k] = relu(sum_j pooled[k][j]*w1[o][j] + b1[o])
  for (int oc = tid; oc < 16 * KSORT; oc += 256) {
    const int o = oc / KSORT, kk = oc % KSORT;
    const float* wr = w1 + o * DFEAT;
    const float* pr = pooled + kk * DFEAT;
    float s = 0.f;
    for (int j = 0; j < DFEAT; ++j) s += pr[j] * wr[j];
    c1[o * KSORT + kk] = fmaxf(s + b1[o], 0.f);
  }
  __syncthreads();

  // maxpool(2,2): (16,30) -> (16,15)
  if (tid < 16 * 15) {
    const int o = tid / 15, t2 = tid % 15;
    pp[tid] = fmaxf(c1[o * KSORT + 2 * t2], c1[o * KSORT + 2 * t2 + 1]);
  }
  __syncthreads();

  // conv2: (32,11) = conv1d(16->32, k=5) + relu
  if (tid < 32 * 11) {
    const int o = tid / 11, t2 = tid % 11;
    float s = 0.f;
    for (int i2 = 0; i2 < 16; ++i2)
      for (int j = 0; j < 5; ++j)
        s += pp[i2 * 15 + t2 + j] * w2[(o * 16 + i2) * 5 + j];
    c2[tid] = fmaxf(s + b2[o], 0.f);     // flat index == o*11+t (matches reshape)
  }
  __syncthreads();

  // lin1 (352->128) + relu
  if (tid < 128) {
    const float* wr = lw1 + tid * 352;
    float s = 0.f;
    for (int i2 = 0; i2 < 352; ++i2) s += c2[i2] * wr[i2];
    hh[tid] = fmaxf(s + lb1[tid], 0.f);
  }
  __syncthreads();

  // lin2 (128->1)
  if (tid == 0) {
    float s = 0.f;
    for (int q = 0; q < 128; ++q) s += hh[q] * lw2[q];
    out[g] = s + lb2[0];
  }
}

// ---------------- launch ----------------
extern "C" void kernel_launch(void* const* d_in, const int* in_sizes, int n_in,
                              void* d_out, int out_size, void* d_ws, size_t ws_size,
                              hipStream_t stream) {
  (void)in_sizes; (void)n_in; (void)out_size; (void)ws_size;
  const int*   z       = (const int*)d_in[0];
  const int*   src     = (const int*)d_in[1];
  const int*   dst     = (const int*)d_in[2];
  const float* z_table = (const float*)d_in[3];
  const float* biases  = (const float*)d_in[4];
  const float* conv1_w = (const float*)d_in[5];
  const float* conv1_b = (const float*)d_in[6];
  const float* conv2_w = (const float*)d_in[7];
  const float* conv2_b = (const float*)d_in[8];
  const float* lin1_w  = (const float*)d_in[9];
  const float* lin1_b  = (const float*)d_in[10];
  const float* lin2_w  = (const float*)d_in[11];
  const float* lin2_b  = (const float*)d_in[12];
  float* out = (float*)d_out;

  unsigned char* ws = (unsigned char*)d_ws;
  unsigned* deg_out = (unsigned*)ws;              // N u32
  unsigned* deg_in  = deg_out + NN;               // N u32
  float* norm_src   = (float*)(deg_in + NN);      // N f32
  float* norm_dst   = norm_src + NN;              // N f32
  float* x          = (float*)(ws + ((size_t)4 * NN * 4 + 255) / 256 * 256); // N*384 f32

  k_zero<<<(2 * NN + 255) / 256, 256, 0, stream>>>(deg_out, 2 * NN);
  k_deg<<<(EDGES + 255) / 256, 256, 0, stream>>>(src, dst, deg_out, deg_in);
  k_norm<<<(NN + 255) / 256, 256, 0, stream>>>(deg_out, deg_in, norm_src, norm_dst);

  const size_t shmem = (size_t)(256 * ADJ_STRIDE + 2 * 128 * H_STRIDE) * 2;  // 270336B
  (void)hipFuncSetAttribute((const void*)k_gconv,
                            hipFuncAttributeMaxDynamicSharedMemorySize, (int)shmem);
  k_gconv<<<B_GRAPHS, 256, shmem, stream>>>(z, src, dst, z_table, biases,
                                            norm_src, norm_dst, x);
  k_head<<<B_GRAPHS, 256, 0, stream>>>(x, conv1_w, conv1_b, conv2_w, conv2_b,
                                       lin1_w, lin1_b, lin2_w, lin2_b, out);
}
